// GPNNCellText_20641612824996
// MI455X (gfx1250) — matrix operand order repaired and checked by reading, not compiled, not run on hardware
//
#include <hip/hip_runtime.h>
#include <math.h>

typedef __attribute__((ext_vector_type(16))) _Float16 v16h;
typedef __attribute__((ext_vector_type(8)))  _Float16 v8h;
typedef __attribute__((ext_vector_type(8)))  float    v8f;

#define DIM 768
#define NNODE 10
#define ROWS_TOTAL 5120      // B*S*N = 4*128*10
#define BS_TOTAL 512         // B*S
#define KTILES 24            // 768/32
#define NTILES 48            // 768/16
#define EDGE_TILES 7         // ceil(100 edges / 16)

__device__ __forceinline__ float gelu_f(float v) {
  return 0.5f * v * (1.0f + erff(v * 0.70710678118654752f));
}

// A fragment per CDNA5 ISA 16-bit A 16x32 layout:
// g = lane>>4, m = lane&15; halves 0..7 -> K = 8g..8g+7, halves 8..15 -> K = 16+8g..+7
__device__ __forceinline__ v16h load_a_frag(const _Float16* __restrict__ A, int ld,
                                            int mrow, int g, int k0) {
  const _Float16* p = A + mrow * ld + k0 + 8 * g;
  v8h a0 = *(const v8h*)(p);
  v8h a1 = *(const v8h*)(p + 16);
  v16h r;
#pragma unroll
  for (int i = 0; i < 8; ++i) { r[i] = a0[i]; r[i + 8] = a1[i]; }
  return r;
}

// 16x768 (A, row-major f16) times packed 768x768 B; each wave owns 6 column tiles.
// K-outer, FULLY UNROLLED: each A fragment loaded once feeds 6 WMMAs; accumulators
// stay pinned in registers (no loop-carried moves -> no WMMA->VALU hazard NOPs).
__device__ __forceinline__ void gemm16(const _Float16* __restrict__ A,
                                       const _Float16* __restrict__ Bp,
                                       v8f accs[6], int wave, int lane) {
  const int g = lane >> 4, mrow = lane & 15;
  const v8f vzero = {};
#pragma unroll
  for (int c = 0; c < 6; ++c) accs[c] = vzero;
#pragma unroll
  for (int kt = 0; kt < KTILES; ++kt) {
    const v16h a = load_a_frag(A, DIM, mrow, g, kt * 32);
    if (kt + 1 < KTILES)
      __builtin_prefetch(Bp + ((((kt + 1) * NTILES + wave * 6) << 9) + (lane << 4)), 0, 1);
#pragma unroll
    for (int c = 0; c < 6; ++c) {
      const int nt = wave * 6 + c;
      const v16h b = *(const v16h*)(Bp + (((kt * NTILES + nt) << 9) + (lane << 4)));
      accs[c] = __builtin_amdgcn_wmma_f32_16x16x32_f16(false, a, false, b, (short)0,
                                                       accs[c], false, false);
    }
  }
}

// ---------------- small utility kernels ----------------

__global__ __launch_bounds__(256) void cvt_f32_f16_kernel(const float* __restrict__ src,
                                                          _Float16* __restrict__ dst, int n) {
  int idx = blockIdx.x * 256 + threadIdx.x;
  if (idx < n) dst[idx] = (_Float16)src[idx];
}

// Pack a 768x768 fp32 weight (row-major, row = K, col = N) into WMMA B-fragment order.
__global__ __launch_bounds__(256) void packW_kernel(const float* __restrict__ src,
                                                    _Float16* __restrict__ dst) {
  int idx = blockIdx.x * 256 + threadIdx.x;           // 768*768 = 589824 total
  const int h = idx & 15;
  const int lane = (idx >> 4) & 31;
  const int t = idx >> 9;
  const int nt = t % NTILES;
  const int kt = t / NTILES;
  const int kk = kt * 32 + (h & 7) + ((h >> 3) << 4) + ((lane >> 4) << 3);
  const int nn = nt * 16 + (lane & 15);
  dst[idx] = (_Float16)src[kk * DIM + nn];
}

// Row-tile WMMA GEMM, 16 rows per block: out = A[M x 768] @ Wpacked (+bias).
// mode 0: f16 out, no bias. mode 1: f32 out + bias.
__global__ __launch_bounds__(256) void gemm_rm_kernel(const _Float16* __restrict__ A,
                                                      const _Float16* __restrict__ Bp,
                                                      const float* __restrict__ bias,
                                                      _Float16* __restrict__ out16,
                                                      float* __restrict__ out32, int mode) {
  const int tid = threadIdx.x, wave = tid >> 5, lane = tid & 31;
  const int g = lane >> 4, mrow = lane & 15;
  const long row0 = (long)blockIdx.x * 16;
  v8f accs[6];
  gemm16(A + row0 * DIM, Bp, accs, wave, lane);
#pragma unroll
  for (int c = 0; c < 6; ++c) {
    const int n = (wave * 6 + c) * 16 + mrow;
    const float bv = (mode == 1) ? bias[n] : 0.f;
#pragma unroll
    for (int r = 0; r < 8; ++r) {
      const long grow = row0 + r + 8 * g;
      if (mode == 1) out32[grow * DIM + n] = accs[c][r] + bv;
      else           out16[grow * DIM + n] = (_Float16)accs[c][r];
    }
  }
}

// ---------------- fused edge-message kernel: one block per (b,s) ----------------

__global__ __launch_bounds__(256) void edge_kernel(
    const float* __restrict__ x,
    const _Float16* __restrict__ a16, const _Float16* __restrict__ c16,
    const _Float16* __restrict__ xm16,
    const float* __restrict__ b1, const float* __restrict__ b2,
    const float* __restrict__ ln1g, const float* __restrict__ ln1b,
    const float* __restrict__ Wl, const float* __restrict__ bl,
    const float* __restrict__ bm, const float* __restrict__ bmer,
    const float* __restrict__ ln2g, const float* __restrict__ ln2b,
    const float* __restrict__ msc,
    const _Float16* __restrict__ W2p, const _Float16* __restrict__ Wm2p,
    const _Float16* __restrict__ Wmerp,
    float* __restrict__ ew, _Float16* __restrict__ res16) {
  __shared__ _Float16 sh[16 * DIM];   // 24 KB: h tile / m tile (f16)
  __shared__ _Float16 st[16 * DIM];   // 24 KB: GEMM outputs, ef / mv tiles (f16)
  __shared__ float sw[16];
  __shared__ float sxn[NNODE], sen[NNODE];

  const int tid = threadIdx.x, wave = tid >> 5, lane = tid & 31;
  const int g = lane >> 4, mrow = lane & 15;
  const int bs = blockIdx.x;
  const long base = (long)bs * NNODE * DIM;
  float* __restrict__ ewp = ew + base;

  for (int i = tid; i < NNODE * DIM; i += 256) ewp[i] = 0.f;
  __syncthreads();

  for (int t = 0; t < EDGE_TILES; ++t) {
    // h = gelu(a_i + c_j + b1), edge e = 16t + row, i = e/10 (receiver), j = e%10 (sender)
    for (int idx = tid; idx < 16 * DIM; idx += 256) {
      const int row = idx / DIM, k = idx - row * DIM;
      const int e = t * 16 + row;
      const int i = (e < 100) ? (e / 10) : 0;
      const int j = (e < 100) ? (e % 10) : 0;
      const float v = (float)a16[base + i * DIM + k] + (float)c16[base + j * DIM + k] + b1[k];
      sh[idx] = (_Float16)gelu_f(v);
    }
    __syncthreads();

    v8f accs[6];
    // ef_raw = h @ W2
    gemm16(sh, W2p, accs, wave, lane);
#pragma unroll
    for (int c = 0; c < 6; ++c) {
      const int n = (wave * 6 + c) * 16 + mrow;
#pragma unroll
      for (int r = 0; r < 8; ++r) st[(r + 8 * g) * DIM + n] = (_Float16)accs[c][r];
    }
    __syncthreads();

    // ef = gelu(LN(ef_raw + b2)); gate w = sigmoid(ef . Wl + bl)
    {
      const int row = tid >> 4, lr = tid & 15;
      float s = 0.f, ss = 0.f;
      for (int q = 0; q < 48; ++q) {
        const int k = lr + (q << 4);
        const float v = (float)st[row * DIM + k] + b2[k];
        s += v; ss += v * v;
      }
#pragma unroll
      for (int o = 8; o; o >>= 1) { s += __shfl_xor(s, o, 16); ss += __shfl_xor(ss, o, 16); }
      const float mean = s * (1.f / 768.f);
      const float inv  = rsqrtf(fmaxf(ss * (1.f / 768.f) - mean * mean, 0.f) + 1e-12f);
      float wdot = 0.f;
      for (int q = 0; q < 48; ++q) {
        const int k = lr + (q << 4);
        const float v = (float)st[row * DIM + k] + b2[k];
        const float ef = gelu_f((v - mean) * inv * ln1g[k] + ln1b[k]);
        st[row * DIM + k] = (_Float16)ef;
        wdot += ef * Wl[k];
      }
#pragma unroll
      for (int o = 8; o; o >>= 1) wdot += __shfl_xor(wdot, o, 16);
      if (lr == 0) {
        const int e = t * 16 + row;
        sw[row] = (e < 100) ? 1.f / (1.f + expf(-(wdot + bl[0]))) : 0.f;
      }
    }
    __syncthreads();

    // m = ef @ Wm[D:] + xm_j + bm  (GEMM result held in VGPRs across the barrier)
    gemm16(st, Wm2p, accs, wave, lane);
    __syncthreads();   // all waves done reading st
#pragma unroll
    for (int c = 0; c < 6; ++c) {
      const int n = (wave * 6 + c) * 16 + mrow;
      const float bmv = bm[n];
#pragma unroll
      for (int r = 0; r < 8; ++r) {
        const int m = r + 8 * g;
        const int e = t * 16 + m;
        const int j = (e < 100) ? (e % 10) : 0;
        sh[m * DIM + n] = (_Float16)(accs[c][r] + (float)xm16[base + j * DIM + n] + bmv);
      }
    }
    __syncthreads();

    // mv_raw = m @ Wmer
    gemm16(sh, Wmerp, accs, wave, lane);
#pragma unroll
    for (int c = 0; c < 6; ++c) {
      const int n = (wave * 6 + c) * 16 + mrow;
#pragma unroll
      for (int r = 0; r < 8; ++r) st[(r + 8 * g) * DIM + n] = (_Float16)accs[c][r];
    }
    __syncthreads();

    // mv = gelu(LN(mv_raw + bmer)) in place
    {
      const int row = tid >> 4, lr = tid & 15;
      float s = 0.f, ss = 0.f;
      for (int q = 0; q < 48; ++q) {
        const int k = lr + (q << 4);
        const float v = (float)st[row * DIM + k] + bmer[k];
        s += v; ss += v * v;
      }
#pragma unroll
      for (int o = 8; o; o >>= 1) { s += __shfl_xor(s, o, 16); ss += __shfl_xor(ss, o, 16); }
      const float mean = s * (1.f / 768.f);
      const float inv  = rsqrtf(fmaxf(ss * (1.f / 768.f) - mean * mean, 0.f) + 1e-12f);
      for (int q = 0; q < 48; ++q) {
        const int k = lr + (q << 4);
        const float v = (float)st[row * DIM + k] + bmer[k];
        st[row * DIM + k] = (_Float16)gelu_f((v - mean) * inv * ln2g[k] + ln2b[k]);
      }
    }
    __syncthreads();

    // ew[i] += w * mv : each thread owns 3 k-columns, serial over rows -> race free
    for (int kk = 0; kk < 3; ++kk) {
      const int k = tid * 3 + kk;
#pragma unroll
      for (int row = 0; row < 16; ++row) {
        const int e = t * 16 + row;
        if (e < 100) ewp[(e / 10) * DIM + k] += sw[row] * (float)st[row * DIM + k];
      }
    }
    __syncthreads();
  }

  // MessageNorm + residual: res = ew/max(||ew||,1e-12) * ||x|| * scale + x
  if (tid < 16 * NNODE) {
    const int row = tid >> 4, lr = tid & 15;
    float sx = 0.f, se = 0.f;
    for (int q = 0; q < 48; ++q) {
      const int k = lr + (q << 4);
      const float xv = x[base + row * DIM + k];
      const float ev = ewp[row * DIM + k];
      sx += xv * xv; se += ev * ev;
    }
#pragma unroll
    for (int o = 8; o; o >>= 1) { sx += __shfl_xor(sx, o, 16); se += __shfl_xor(se, o, 16); }
    if (lr == 0) { sxn[row] = sqrtf(sx); sen[row] = fmaxf(sqrtf(se), 1e-12f); }
  }
  __syncthreads();
  const float scale = msc[0];
  for (int idx = tid; idx < NNODE * DIM; idx += 256) {
    const int row = idx / DIM;
    const float r = ewp[idx] / sen[row] * sxn[row] * scale + x[base + idx];
    res16[base + idx] = (_Float16)r;
  }
}

// ---------------- GraphNorm ----------------

__global__ __launch_bounds__(256) void stats_kernel(const float* __restrict__ y,
                                                    float* __restrict__ mu,
                                                    float* __restrict__ ms2) {
  __shared__ float r1[256], r2[256];
  const int c = blockIdx.x, tid = threadIdx.x;
  float s1 = 0.f, s2 = 0.f;
  for (int r = tid; r < ROWS_TOTAL; r += 256) {
    const float v = y[(long)r * DIM + c];
    s1 += v; s2 += v * v;
  }
  r1[tid] = s1; r2[tid] = s2;
  __syncthreads();
  for (int o = 128; o; o >>= 1) {
    if (tid < o) { r1[tid] += r1[tid + o]; r2[tid] += r2[tid + o]; }
    __syncthreads();
  }
  if (tid == 0) { mu[c] = r1[0] * (1.f / ROWS_TOTAL); ms2[c] = r2[0] * (1.f / ROWS_TOTAL); }
}

__global__ __launch_bounds__(256) void final_kernel(const float* __restrict__ y,
                                                    const float* __restrict__ mu,
                                                    const float* __restrict__ ms2,
                                                    const float* __restrict__ gnw,
                                                    const float* __restrict__ gnb,
                                                    const float* __restrict__ gnm,
                                                    float* __restrict__ out) {
  const int idx = blockIdx.x * 256 + threadIdx.x;   // ROWS_TOTAL*DIM elements
  const int c = idx % DIM;
  const float a = gnm[c], m = mu[c];
  // var = E[(y - a*mu)^2] = E[y^2] - mu^2*(2a - a^2)   (since mu = E[y])
  const float var = ms2[c] - m * m * (2.f * a - a * a);
  const float yc = y[idx] - a * m;
  out[idx] = gelu_f(gnw[c] * yc * rsqrtf(var + 1e-5f) + gnb[c]);
}

// ---------------- host launcher ----------------

extern "C" void kernel_launch(void* const* d_in, const int* in_sizes, int n_in,
                              void* d_out, int out_size, void* d_ws, size_t ws_size,
                              hipStream_t stream) {
  (void)in_sizes; (void)n_in; (void)out_size; (void)ws_size;
  const float* x    = (const float*)d_in[0];
  const float* W1   = (const float*)d_in[1];
  const float* b1   = (const float*)d_in[2];
  const float* W2   = (const float*)d_in[3];
  const float* b2   = (const float*)d_in[4];
  const float* ln1g = (const float*)d_in[5];
  const float* ln1b = (const float*)d_in[6];
  const float* Wl   = (const float*)d_in[7];
  const float* bl   = (const float*)d_in[8];
  const float* Wm   = (const float*)d_in[9];
  const float* bm   = (const float*)d_in[10];
  const float* Wmer = (const float*)d_in[11];
  const float* bmer = (const float*)d_in[12];
  const float* ln2g = (const float*)d_in[13];
  const float* ln2b = (const float*)d_in[14];
  const float* msc  = (const float*)d_in[15];
  const float* Wn   = (const float*)d_in[16];
  const float* bn   = (const float*)d_in[17];
  const float* gnw  = (const float*)d_in[18];
  const float* gnb  = (const float*)d_in[19];
  const float* gnm  = (const float*)d_in[20];

  char* ws = (char*)d_ws;
  size_t off = 0;
  auto walloc = [&](size_t bytes) -> void* {
    void* p = (void*)(ws + off);
    off += (bytes + 255) & ~(size_t)255;
    return p;
  };
  const size_t actH = (size_t)ROWS_TOTAL * DIM * sizeof(_Float16);   // 7.86 MB
  const size_t wH   = (size_t)DIM * DIM * sizeof(_Float16);          // 1.18 MB
  _Float16* xf    = (_Float16*)walloc(actH);
  _Float16* af    = (_Float16*)walloc(actH);
  _Float16* cf    = (_Float16*)walloc(actH);
  _Float16* xmf   = (_Float16*)walloc(actH);
  _Float16* resf  = (_Float16*)walloc(actH);
  _Float16* W1aP  = (_Float16*)walloc(wH);
  _Float16* W1cP  = (_Float16*)walloc(wH);
  _Float16* W2P   = (_Float16*)walloc(wH);
  _Float16* Wm1P  = (_Float16*)walloc(wH);
  _Float16* Wm2P  = (_Float16*)walloc(wH);
  _Float16* WmerP = (_Float16*)walloc(wH);
  _Float16* WnP   = (_Float16*)walloc(wH);
  float* ewbuf = (float*)walloc((size_t)ROWS_TOTAL * DIM * sizeof(float));  // 15.7 MB
  float* ybuf  = (float*)walloc((size_t)ROWS_TOTAL * DIM * sizeof(float));  // 15.7 MB
  float* mu    = (float*)walloc(DIM * sizeof(float));
  float* ms2   = (float*)walloc(DIM * sizeof(float));

  const int nAct = ROWS_TOTAL * DIM;                 // 3,932,160
  cvt_f32_f16_kernel<<<(nAct + 255) / 256, 256, 0, stream>>>(x, xf, nAct);

  const int packGrid = (DIM * DIM) / 256;            // 2304
  packW_kernel<<<packGrid, 256, 0, stream>>>(W1,            W1aP);
  packW_kernel<<<packGrid, 256, 0, stream>>>(W1 + DIM*DIM,  W1cP);
  packW_kernel<<<packGrid, 256, 0, stream>>>(W2,            W2P);
  packW_kernel<<<packGrid, 256, 0, stream>>>(Wm,            Wm1P);
  packW_kernel<<<packGrid, 256, 0, stream>>>(Wm + DIM*DIM,  Wm2P);
  packW_kernel<<<packGrid, 256, 0, stream>>>(Wmer,          WmerP);
  packW_kernel<<<packGrid, 256, 0, stream>>>(Wn,            WnP);

  const int gemmGrid = ROWS_TOTAL / 16;              // 320
  gemm_rm_kernel<<<gemmGrid, 256, 0, stream>>>(xf, W1aP, nullptr, af,  nullptr, 0);
  gemm_rm_kernel<<<gemmGrid, 256, 0, stream>>>(xf, W1cP, nullptr, cf,  nullptr, 0);
  gemm_rm_kernel<<<gemmGrid, 256, 0, stream>>>(xf, Wm1P, nullptr, xmf, nullptr, 0);

  edge_kernel<<<BS_TOTAL, 256, 0, stream>>>(x, af, cf, xmf, b1, b2, ln1g, ln1b,
                                            Wl, bl, bm, bmer, ln2g, ln2b, msc,
                                            W2P, Wm2P, WmerP, ewbuf, resf);

  gemm_rm_kernel<<<gemmGrid, 256, 0, stream>>>(resf, WnP, bn, nullptr, ybuf, 1);

  stats_kernel<<<DIM, 256, 0, stream>>>(ybuf, mu, ms2);
  final_kernel<<<nAct / 256, 256, 0, stream>>>(ybuf, mu, ms2, gnw, gnb, gnm,
                                               (float*)d_out);
}
